// Decoder_65335042506813
// MI455X (gfx1250) — matrix operand order repaired and checked by reading, not compiled, not run on hardware
//
#include <hip/hip_runtime.h>
#include <cstddef>

// ---------------------------------------------------------------------------
// Problem constants (from the reference)
// ---------------------------------------------------------------------------
#define NRES 4096
#define KNB  32
#define DCH  256
#define NH   8
#define HKD  32
#define PCH  64
#define NATM 14
#define NLAY 2
#define NPQ_ 4
#define NPV_ 8

// ---------------------------------------------------------------------------
// WMMA types (CDNA5 / gfx1250, wave32)
// ---------------------------------------------------------------------------
typedef __attribute__((ext_vector_type(16))) __bf16    bf16x16;
typedef __attribute__((ext_vector_type(8)))  float     f32x8;
typedef __attribute__((ext_vector_type(8)))  unsigned  u32x8;

// ---------------------------------------------------------------------------
// Small device helpers
// ---------------------------------------------------------------------------
__device__ inline float wred_sum(float v) {
  for (int m = 16; m > 0; m >>= 1) v += __shfl_xor(v, m, 32);
  return v;
}
__device__ inline float wred_max(float v) {
  for (int m = 16; m > 0; m >>= 1) v = fmaxf(v, __shfl_xor(v, m, 32));
  return v;
}
__device__ inline float gelu_f(float x) {
  const float k0 = 0.7978845608028654f, k1 = 0.044715f;
  float x3 = x * x * x;
  return 0.5f * x * (1.0f + tanhf(k0 * (x + k1 * x3)));
}
__device__ inline float softplus_f(float x) {
  return x > 20.0f ? x : log1pf(expf(x));
}
__device__ inline unsigned pack_bf16(float a, float b) {
  union { __bf16 h[2]; unsigned u; } p;
  p.h[0] = (__bf16)a;       // -> v_cvt_pk_bf16_f32
  p.h[1] = (__bf16)b;
  return p.u;
}

// ---------------------------------------------------------------------------
// Generic guarded GEMM: C[M,N] = epi( A[M,K] * B[K,N] )
// fp32 in/out, bf16 WMMA compute (v_wmma_f32_16x16x32_bf16).
// Tile: 64x64 per block (128 threads = 4 waves), each wave 32x32 (2x2 frags).
// LDS tiles hold packed bf16 pairs (one u32 = two consecutive-K elements):
//   AsU[row][cp]  : A[row][2cp],A[row][2cp+1]      stride 18 u32 (pad)
//   BsU[col][kp]  : B[2kp][col],B[2kp+1][col]      stride 17 u32 (pad)
// Fast path (full tile, 16B-aligned rows): float4 loads, no guards.
// Ragged path: clamped branchless scalar loads (no divergent exec masking).
// Epilogue: v = acc (+bias[col]); if act v=gelu(v); if mul v*=mul[r,c];
//           if beta v += C[r,c];  C[r,c] = v.
// ---------------------------------------------------------------------------
#define TM 64
#define TN 64
#define TK 32
#define LDAU 18   /* u32 stride per A row (16 pairs + 2 pad) */
#define LDBU 17   /* u32 stride per B col (16 pairs + 1 pad) */

__global__ __launch_bounds__(128)
void gemm_bf16_wmma(int M, int N, int K,
                    const float* __restrict__ A, int lda,
                    const float* __restrict__ B, int ldb,
                    float* __restrict__ C, int ldc,
                    const float* __restrict__ bias,
                    const float* __restrict__ mulsrc,
                    int beta, int act)
{
  __shared__ unsigned AsU[TM * LDAU];
  __shared__ unsigned BsU[TN * LDBU];

  const int tid  = threadIdx.x;
  const int lane = tid & 31;
  const int wave = tid >> 5;
  const int wm = (wave >> 1) * 32;   // wave row offset inside tile
  const int wn = (wave & 1)  * 32;   // wave col offset inside tile
  const int m0 = blockIdx.y * TM;
  const int n0 = blockIdx.x * TN;

  const bool fullMN = (m0 + TM <= M) && (n0 + TN <= N);
  const bool alnA = (lda & 3) == 0;
  const bool alnB = (ldb & 3) == 0;

  f32x8 acc[2][2];
  for (int i = 0; i < 2; ++i)
    for (int j = 0; j < 2; ++j)
      for (int e = 0; e < 8; ++e) acc[i][j][e] = 0.0f;

  const int r  = lane & 15;
  const int hi = lane >> 4;

  for (int k0 = 0; k0 < K; k0 += TK) {
    const bool fullK = (k0 + TK <= K);

    // ---------------- stage A tile (64 x 32) ----------------
    if (fullMN && fullK && alnA) {
#pragma unroll
      for (int j = 0; j < 4; ++j) {
        int i   = tid + 128 * j;          // 512 float4 units
        int row = i >> 3, c4 = (i & 7) * 4;
        const float4 f =
            *(const float4*)&A[(size_t)(m0 + row) * lda + k0 + c4];
        uint2 pk;
        pk.x = pack_bf16(f.x, f.y);
        pk.y = pack_bf16(f.z, f.w);
        *(uint2*)&AsU[row * LDAU + (c4 >> 1)] = pk;
      }
    } else {
      for (int e = tid; e < TM * (TK / 2); e += 128) {
        int row = e >> 4, cp = e & 15;
        int gr = m0 + row, gk = k0 + 2 * cp;
        int cr  = gr < M ? gr : (M - 1);
        int ck0 = gk < K ? gk : (K - 1);
        int ck1 = (gk + 1) < K ? (gk + 1) : (K - 1);
        float f0 = A[(size_t)cr * lda + ck0];
        float f1 = A[(size_t)cr * lda + ck1];
        f0 = (gr < M && gk < K) ? f0 : 0.0f;
        f1 = (gr < M && (gk + 1) < K) ? f1 : 0.0f;
        AsU[row * LDAU + cp] = pack_bf16(f0, f1);
      }
    }
    // ---------------- stage B tile (32 x 64) ----------------
    if (fullMN && fullK && alnB) {
#pragma unroll
      for (int j = 0; j < 2; ++j) {
        int i  = tid + 128 * j;           // 256 pair-of-float4 units
        int kp = i >> 4, c4 = (i & 15) * 4;
        const float4 r0 =
            *(const float4*)&B[(size_t)(k0 + 2 * kp) * ldb + n0 + c4];
        const float4 r1 =
            *(const float4*)&B[(size_t)(k0 + 2 * kp + 1) * ldb + n0 + c4];
        BsU[(c4 + 0) * LDBU + kp] = pack_bf16(r0.x, r1.x);
        BsU[(c4 + 1) * LDBU + kp] = pack_bf16(r0.y, r1.y);
        BsU[(c4 + 2) * LDBU + kp] = pack_bf16(r0.z, r1.z);
        BsU[(c4 + 3) * LDBU + kp] = pack_bf16(r0.w, r1.w);
      }
    } else {
      for (int e = tid; e < TN * (TK / 2); e += 128) {
        int kp = e >> 6, c = e & 63;
        int gc = n0 + c, gk = k0 + 2 * kp;
        int cc  = gc < N ? gc : (N - 1);
        int ck0 = gk < K ? gk : (K - 1);
        int ck1 = (gk + 1) < K ? (gk + 1) : (K - 1);
        float f0 = B[(size_t)ck0 * ldb + cc];
        float f1 = B[(size_t)ck1 * ldb + cc];
        f0 = (gc < N && gk < K) ? f0 : 0.0f;
        f1 = (gc < N && (gk + 1) < K) ? f1 : 0.0f;
        BsU[c * LDBU + kp] = pack_bf16(f0, f1);
      }
    }
    // ---- prefetch next K-tile (global_prefetch_b8) ----
    if (k0 + TK < K) {
      int pr = m0 + (tid >> 1);
      if (pr < M) __builtin_prefetch(&A[(size_t)pr * lda + (k0 + TK)], 0, 1);
      int br = k0 + TK + (tid & 31);
      if (br < K) __builtin_prefetch(&B[(size_t)br * ldb + n0], 0, 1);
    }
    __syncthreads();

    // ---- load fragments (ISA 7.12.2 layouts, pairs = one u32) ----
    bf16x16 afr[2], bfr[2];
    for (int mi = 0; mi < 2; ++mi) {
      int mo = wm + mi * 16;
      u32x8 au;
#pragma unroll
      for (int v2 = 0; v2 < 8; ++v2) {
        int cp = (v2 < 4 ? v2 : v2 + 4) + hi * 4;
        au[v2] = AsU[(mo + r) * LDAU + cp];
      }
      afr[mi] = __builtin_bit_cast(bf16x16, au);
    }
    for (int ni = 0; ni < 2; ++ni) {
      int no = wn + ni * 16;
      u32x8 bu;
#pragma unroll
      for (int v2 = 0; v2 < 8; ++v2) {
        int kp = (v2 < 4 ? v2 : v2 + 4) + hi * 4;
        bu[v2] = BsU[(no + r) * LDBU + kp];
      }
      bfr[ni] = __builtin_bit_cast(bf16x16, bu);
    }
#pragma unroll
    for (int mi = 0; mi < 2; ++mi)
#pragma unroll
      for (int ni = 0; ni < 2; ++ni)
        acc[mi][ni] = __builtin_amdgcn_wmma_f32_16x16x32_bf16(
            false, afr[mi], false, bfr[ni], (short)0, acc[mi][ni], false, false);
    __syncthreads();
  }

  // ---- epilogue ----
  for (int mi = 0; mi < 2; ++mi)
    for (int ni = 0; ni < 2; ++ni) {
#pragma unroll
      for (int v2 = 0; v2 < 8; ++v2) {
        int row = m0 + wm + mi * 16 + v2 + hi * 8;
        int col = n0 + wn + ni * 16 + r;
        if (row < M && col < N) {
          float v = acc[mi][ni][v2];
          if (bias)   v += bias[col];
          if (act)    v  = gelu_f(v);
          if (mulsrc) v *= mulsrc[(size_t)row * ldc + col];
          float* cp = &C[(size_t)row * ldc + col];
          if (beta)   v += *cp;
          *cp = v;
        }
      }
    }
}

// ---------------------------------------------------------------------------
// LayerNorm over D=256 (one block / row, 256 threads)
// ---------------------------------------------------------------------------
__global__ __launch_bounds__(256)
void ln_kernel(const float* __restrict__ in, float* __restrict__ out)
{
  int n = blockIdx.x, t = threadIdx.x;
  __shared__ float red[8];
  float x = in[(size_t)n * DCH + t];
  float s = wred_sum(x);
  if ((t & 31) == 0) red[t >> 5] = s;
  __syncthreads();
  float tot = 0.f;
  for (int i = 0; i < 8; ++i) tot += red[i];
  float m = tot * (1.0f / DCH);
  float d = x - m;
  __syncthreads();
  float s2 = wred_sum(d * d);
  if ((t & 31) == 0) red[t >> 5] = s2;
  __syncthreads();
  float var = 0.f;
  for (int i = 0; i < 8; ++i) var += red[i];
  var *= (1.0f / DCH);
  out[(size_t)n * DCH + t] = d * rsqrtf(var + 1e-5f);
}

// ---------------------------------------------------------------------------
// Frames: R (N,3,3 cols=e1,e2,e3), t=CA; mode0 -> feat[266], mode1 -> ll[42]
// ---------------------------------------------------------------------------
__global__ __launch_bounds__(64)
void frames_kernel(const float* __restrict__ pos, float* __restrict__ Rb,
                   float* __restrict__ tb, float* __restrict__ outf, int mode)
{
  int n = blockIdx.x, t = threadIdx.x;
  __shared__ float R[9], tt[3], lp[42], dist[14];
  if (t == 0) {
    const float* pr = pos + (size_t)n * 42;
    float Nv[3] = {pr[0], pr[1], pr[2]};
    float ca[3] = {pr[3], pr[4], pr[5]};
    float Cv[3] = {pr[6], pr[7], pr[8]};
    float e1[3], u[3], e2[3], e3[3];
    float s = 1e-12f;
    for (int d = 0; d < 3; ++d) { e1[d] = Cv[d] - ca[d]; s += e1[d] * e1[d]; }
    float inv = 1.0f / sqrtf(s);
    for (int d = 0; d < 3; ++d) e1[d] *= inv;
    float du = 0.f;
    for (int d = 0; d < 3; ++d) { u[d] = Nv[d] - ca[d]; }
    for (int d = 0; d < 3; ++d) du += u[d] * e1[d];
    s = 1e-12f;
    for (int d = 0; d < 3; ++d) { e2[d] = u[d] - du * e1[d]; s += e2[d] * e2[d]; }
    inv = 1.0f / sqrtf(s);
    for (int d = 0; d < 3; ++d) e2[d] *= inv;
    e3[0] = e1[1] * e2[2] - e1[2] * e2[1];
    e3[1] = e1[2] * e2[0] - e1[0] * e2[2];
    e3[2] = e1[0] * e2[1] - e1[1] * e2[0];
    for (int d = 0; d < 3; ++d) {
      R[d * 3 + 0] = e1[d]; R[d * 3 + 1] = e2[d]; R[d * 3 + 2] = e3[d];
      tt[d] = ca[d];
    }
  }
  __syncthreads();
  if (t < 14) {
    float s = 1e-12f;
    for (int c = 0; c < 3; ++c) {
      float acc = 0.f;
      for (int d = 0; d < 3; ++d)
        acc += (pos[(size_t)n * 42 + t * 3 + d] - tt[d]) * R[d * 3 + c];
      lp[t * 3 + c] = acc; s += acc * acc;
    }
    dist[t] = sqrtf(s);
  }
  __syncthreads();
  if (t < 9) Rb[(size_t)n * 9 + t] = R[t];
  if (t < 3) tb[(size_t)n * 3 + t] = tt[t];
  if (mode == 0) {
    if (t < 14) {
      float dn = dist[t];
      for (int c = 0; c < 3; ++c)
        outf[(size_t)n * 266 + t * 3 + c] = lp[t * 3 + c] / dn;
      for (int b = 0; b < 16; ++b) {
        float cen = 10.0f * b / 15.0f;
        float e = (dist[t] - cen) * (1.0f / 0.625f);
        outf[(size_t)n * 266 + 42 + t * 16 + b] = expf(-e * e);
      }
    }
  } else {
    if (t < 42) outf[(size_t)n * 42 + t] = lp[t];
  }
}

// ---------------------------------------------------------------------------
// Pair features -> pair[n,k,64]; one-hot @ wp collapses to a row lookup.
// pf = [rbf16 | nrel3 | onehot65 | chain1] @ wp(85,64)
// ---------------------------------------------------------------------------
__global__ __launch_bounds__(256)
void pair_kernel(const float* __restrict__ Rb, const float* __restrict__ tb,
                 const int* __restrict__ nbr, const int* __restrict__ resi,
                 const int* __restrict__ chain, const float* __restrict__ wp,
                 float* __restrict__ pairb)
{
  int n = blockIdx.x, t = threadIdx.x;
  int k = t >> 3, c0 = (t & 7) * 8;
  __shared__ float R[9], tt[3];
  if (t < 9) R[t] = Rb[(size_t)n * 9 + t];
  if (t < 3) tt[t] = tb[(size_t)n * 3 + t];
  __syncthreads();
  int nb = nbr[n * KNB + k];
  float rel[3]; float s = 1e-12f;
  for (int c = 0; c < 3; ++c) {
    float acc = 0.f;
    for (int d = 0; d < 3; ++d)
      acc += (tb[(size_t)nb * 3 + d] - tt[d]) * R[d * 3 + c];
    rel[c] = acc; s += acc * acc;
  }
  float dist = sqrtf(s);
  float rb[16];
  for (int b = 0; b < 16; ++b) {
    float e = (dist - 10.0f * b / 15.0f) * (1.0f / 0.625f);
    rb[b] = expf(-e * e);
  }
  int dr = resi[nb] - resi[n];
  dr = dr < -32 ? -32 : (dr > 32 ? 32 : dr);
  int rp = dr + 32;
  float ce = (chain[nb] == chain[n]) ? 1.0f : 0.0f;
  float* outp = pairb + ((size_t)n * KNB + k) * PCH;
  for (int cc = c0; cc < c0 + 8; ++cc) {
    float acc = wp[(19 + rp) * PCH + cc] + ce * wp[84 * PCH + cc];
    for (int b = 0; b < 16; ++b) acc += rb[b] * wp[b * PCH + cc];
    for (int j = 0; j < 3; ++j) acc += (rel[j] / dist) * wp[(16 + j) * PCH + cc];
    outp[cc] = acc;
  }
}

// ---------------------------------------------------------------------------
// Point transform: local -> global for qp(96), kp(96), vp(192)
// ---------------------------------------------------------------------------
__global__ __launch_bounds__(192)
void ptrans_kernel(const float* __restrict__ Rb, const float* __restrict__ tb,
                   const float* __restrict__ qp, const float* __restrict__ kp,
                   const float* __restrict__ vp, float* __restrict__ qpg,
                   float* __restrict__ kpg, float* __restrict__ vpg)
{
  int n = blockIdx.x, t = threadIdx.x;
  __shared__ float R[9], tt[3];
  if (t < 9) R[t] = Rb[(size_t)n * 9 + t];
  if (t < 3) tt[t] = tb[(size_t)n * 3 + t];
  __syncthreads();
  if (t < 96) {
    int d = t % 3, base = (t / 3) * 3;
    float a = 0.f, b = 0.f;
    for (int c = 0; c < 3; ++c) {
      a += R[d * 3 + c] * qp[(size_t)n * 96 + base + c];
      b += R[d * 3 + c] * kp[(size_t)n * 96 + base + c];
    }
    qpg[(size_t)n * 96 + t] = a + tt[d];
    kpg[(size_t)n * 96 + t] = b + tt[d];
  }
  {
    int d = t % 3, base = (t / 3) * 3;
    float a = 0.f;
    for (int c = 0; c < 3; ++c)
      a += R[d * 3 + c] * vp[(size_t)n * 192 + base + c];
    vpg[(size_t)n * 192 + t] = a + tt[d];
  }
}

// ---------------------------------------------------------------------------
// IPA attention (block per residue, wave per head, K=32 lanes). Gathers hit L2.
// Output: attcat[n,1024] = [out_s 256 | out_pr 512 | op 192 | |op| 64]
// ---------------------------------------------------------------------------
__global__ __launch_bounds__(256)
void attn_kernel(const int* __restrict__ nbr, const float* __restrict__ qb,
                 const float* __restrict__ kb, const float* __restrict__ vb,
                 const float* __restrict__ qpg, const float* __restrict__ kpg,
                 const float* __restrict__ vpg, const float* __restrict__ pairb,
                 const float* __restrict__ wb, const float* __restrict__ gm,
                 const float* __restrict__ Rb, const float* __restrict__ tb,
                 float* __restrict__ att)
{
  int n = blockIdx.x, t = threadIdx.x;
  int h = t >> 5, lane = t & 31;
  __shared__ int   nbL[32];
  __shared__ float aL[256];
  __shared__ float opgL[192], opL[192];
  __shared__ float R[9], tt[3];
  if (t < 32) nbL[t] = nbr[n * KNB + t];
  if (t < 9)  R[t]  = Rb[(size_t)n * 9 + t];
  if (t < 3)  tt[t] = tb[(size_t)n * 3 + t];
  __syncthreads();

  int kk = lane;
  int nb = nbL[kk];
  float qk = 0.f;
  for (int d = 0; d < HKD; ++d)
    qk += qb[(size_t)n * DCH + h * HKD + d] * kb[(size_t)nb * DCH + h * HKD + d];
  float bias = 0.f;
  for (int p = 0; p < PCH; ++p)
    bias += pairb[((size_t)n * KNB + kk) * PCH + p] * wb[p * NH + h];
  float d2 = 0.f;
  for (int p = 0; p < NPQ_; ++p)
    for (int c = 0; c < 3; ++c) {
      float df = qpg[((size_t)n * NH + h) * 12 + p * 3 + c]
               - kpg[((size_t)nb * NH + h) * 12 + p * 3 + c];
      d2 += df * df;
    }
  const float wc = 0.2357022603955158f;            // sqrt(2/(9*NPQ))
  const float inv_sqrt_kd = 0.17677669529663687f;  // 1/sqrt(32)
  float sp = softplus_f(gm[h]);
  float logit = qk * inv_sqrt_kd + bias - 0.5f * wc * sp * d2;

  float m = wred_max(logit);
  float e = expf(logit - m);
  float ss = wred_sum(e);
  float a = e / ss;
  aL[h * 32 + kk] = a;
  __syncthreads();

  // out_s: lane = d
  {
    float acc = 0.f;
    for (int j = 0; j < 32; ++j)
      acc += aL[h * 32 + j] * vb[(size_t)nbL[j] * DCH + h * HKD + lane];
    att[(size_t)n * 1024 + h * HKD + lane] = acc;
  }
  // out_pr: 64 channels
  for (int it = 0; it < 2; ++it) {
    int pch = lane + it * 32;
    float acc = 0.f;
    for (int j = 0; j < 32; ++j)
      acc += aL[h * 32 + j] * pairb[((size_t)n * KNB + j) * PCH + pch];
    att[(size_t)n * 1024 + 256 + h * PCH + pch] = acc;
  }
  // opg: 24 = NPV*3 items, then transform to local frame
  if (lane < 24) {
    float acc = 0.f;
    for (int j = 0; j < 32; ++j)
      acc += aL[h * 32 + j] * vpg[((size_t)nbL[j] * NH + h) * 24 + lane];
    int c = lane % 3;
    opgL[h * 24 + lane] = acc - tt[c];
  }
  __syncthreads();
  if (lane < 24) {
    int pv = lane / 3, c = lane % 3;
    float acc = 0.f;
    for (int d = 0; d < 3; ++d)
      acc += R[d * 3 + c] * opgL[h * 24 + pv * 3 + d];
    opL[h * 24 + lane] = acc;
    att[(size_t)n * 1024 + 768 + h * 24 + lane] = acc;
  }
  __syncthreads();
  if (lane < 8) {
    float s = 1e-12f;
    for (int c = 0; c < 3; ++c) {
      float v = opL[h * 24 + lane * 3 + c];
      s += v * v;
    }
    att[(size_t)n * 1024 + 960 + h * 8 + lane] = sqrtf(s);
  }
}

// ---------------------------------------------------------------------------
// Position update: pc = R @ ((pc-t)@R + upd)^T + t  (in place via LDS stage)
// ---------------------------------------------------------------------------
__global__ __launch_bounds__(64)
void posupd_kernel(const float* __restrict__ Rb, const float* __restrict__ tb,
                   const float* __restrict__ upd, float* __restrict__ posb)
{
  int n = blockIdx.x, t = threadIdx.x;
  __shared__ float R[9], tt[3], pl[42];
  if (t < 9)  R[t]  = Rb[(size_t)n * 9 + t];
  if (t < 3)  tt[t] = tb[(size_t)n * 3 + t];
  if (t < 42) pl[t] = posb[(size_t)n * 42 + t];
  __syncthreads();
  if (t < 42) {
    int a = t / 3, d = t % 3;
    float out = tt[d];
    for (int c = 0; c < 3; ++c) {
      float lp2 = upd[(size_t)n * 42 + a * 3 + c];
      for (int dd = 0; dd < 3; ++dd)
        lp2 += (pl[a * 3 + dd] - tt[dd]) * R[dd * 3 + c];
      out += R[d * 3 + c] * lp2;
    }
    posb[(size_t)n * 42 + t] = out;
  }
}

// ---------------------------------------------------------------------------
// Final: log_softmax(aa), argmax/amask, build feat2[346]
// ---------------------------------------------------------------------------
__global__ __launch_bounds__(32)
void aa_kernel(const float* __restrict__ aab, const float* __restrict__ x2,
               const float* __restrict__ ll, float* __restrict__ feat2,
               int* __restrict__ aaty, float* __restrict__ out_aa,
               float* __restrict__ out_mask)
{
  int n = blockIdx.x, lane = threadIdx.x;
  float logit = lane < 20 ? aab[(size_t)n * 20 + lane] : -3.4e38f;
  float m = wred_max(logit);
  float e = lane < 20 ? expf(logit - m) : 0.f;
  float s = wred_sum(e);
  if (lane < 20) out_aa[(size_t)n * 20 + lane] = logit - m - logf(s);
  float vmax = logit; int idx = lane < 20 ? lane : 31;
  for (int msk = 16; msk > 0; msk >>= 1) {
    float ov = __shfl_xor(vmax, msk, 32);
    int   oi = __shfl_xor(idx,  msk, 32);
    if (ov > vmax || (ov == vmax && oi < idx)) { vmax = ov; idx = oi; }
  }
  int at = idx;
  if (lane == 0) aaty[n] = at;
  int cnt = 4 + (at * 7) % 11;
  float* f = feat2 + (size_t)n * 346;
  if (lane < 20) f[lane] = (lane == at) ? 1.0f : 0.0f;
  if (lane < 14) {
    float mk = (lane < cnt) ? 1.0f : 0.0f;
    out_mask[(size_t)n * 14 + lane] = mk;
    f[20 + lane] = mk;
    float s2 = 0.f;
    for (int c = 0; c < 3; ++c) {
      float v = ll[(size_t)n * 42 + lane * 3 + c];
      s2 += v * v;
    }
    f[34 + lane] = s2;
  }
  for (int j = lane; j < 42; j += 32) {
    int a = j / 3;
    float s2 = 1e-12f;
    for (int c = 0; c < 3; ++c) {
      float v = ll[(size_t)n * 42 + a * 3 + c];
      s2 += v * v;
    }
    f[48 + j] = ll[(size_t)n * 42 + j] / sqrtf(s2);
  }
  for (int j = lane; j < 256; j += 32)
    f[90 + j] = x2[(size_t)n * DCH + j];
}

__global__ __launch_bounds__(32)
void p4_kernel(const float* __restrict__ ll, float* __restrict__ pb)
{
  int n = blockIdx.x, t = threadIdx.x;
  if (t < 12) pb[(size_t)n * 42 + t] = ll[(size_t)n * 42 + t];
}

__global__ __launch_bounds__(128)
void zcat_kernel(const float* __restrict__ locF, const float* __restrict__ pb,
                 float* __restrict__ z)
{
  int n = blockIdx.x, t = threadIdx.x;
  for (int j = t; j < 298; j += 128)
    z[(size_t)n * 298 + j] =
        j < 256 ? locF[(size_t)n * DCH + j] : pb[(size_t)n * 42 + (j - 256)];
}

__global__ __launch_bounds__(64)
void posout_kernel(const float* __restrict__ Rb, const float* __restrict__ tb,
                   const float* __restrict__ pb, float* __restrict__ out)
{
  int n = blockIdx.x, t = threadIdx.x;
  __shared__ float R[9], tt[3];
  if (t < 9) R[t] = Rb[(size_t)n * 9 + t];
  if (t < 3) tt[t] = tb[(size_t)n * 3 + t];
  __syncthreads();
  if (t < 42) {
    int a = t / 3, d = t % 3;
    float acc = tt[d];
    for (int c = 0; c < 3; ++c)
      acc += R[d * 3 + c] * pb[(size_t)n * 42 + a * 3 + c];
    out[(size_t)n * 42 + t] = acc;
  }
}

// ---------------------------------------------------------------------------
// Host-side launcher
// ---------------------------------------------------------------------------
extern "C" void kernel_launch(void* const* d_in, const int* in_sizes, int n_in,
                              void* d_out, int out_size, void* d_ws, size_t ws_size,
                              hipStream_t stream)
{
  (void)in_sizes; (void)n_in; (void)out_size; (void)ws_size;
  const int N = NRES;

  const float* in_local = (const float*)d_in[0];
  const float* in_pos   = (const float*)d_in[1];
  const int*   nbr      = (const int*)d_in[2];
  const int*   resi     = (const int*)d_in[3];
  const int*   chain    = (const int*)d_in[4];
  const float* w_feat   = (const float*)d_in[5];
  const float* w_pair   = (const float*)d_in[6];
  const float* w_q      = (const float*)d_in[7];
  const float* w_k      = (const float*)d_in[8];
  const float* w_v      = (const float*)d_in[9];
  const float* w_qp     = (const float*)d_in[10];
  const float* w_kp     = (const float*)d_in[11];
  const float* w_vp     = (const float*)d_in[12];
  const float* w_bias   = (const float*)d_in[13];
  const float* gamma    = (const float*)d_in[14];
  const float* w_ao     = (const float*)d_in[15];
  const float* w_g      = (const float*)d_in[16];
  const float* w_vm     = (const float*)d_in[17];
  const float* w_mo     = (const float*)d_in[18];
  const float* w_pu     = (const float*)d_in[19];
  const float* w_aa     = (const float*)d_in[20];
  const float* w_m1     = (const float*)d_in[21];
  const float* b_m1     = (const float*)d_in[22];
  const float* w_m2     = (const float*)d_in[23];
  const float* b_m2     = (const float*)d_in[24];
  const float* w_init   = (const float*)d_in[25];
  const float* b_init   = (const float*)d_in[26];
  const float* w_ug     = (const float*)d_in[27];
  const float* w_uv     = (const float*)d_in[28];
  const float* w_uo     = (const float*)d_in[29];

  float* out = (float*)d_out;
  float* out_aa   = out;                   // N*20
  float* out_pos  = out + (size_t)N * 20;  // N*42
  float* out_mask = out + (size_t)N * 62;  // N*14

  // ---- carve workspace ----
  float* ws = (float*)d_ws;
  size_t o = 0;
  auto alloc = [&](size_t nf) { float* p = ws + o; o += nf; return p; };
  float* Rb    = alloc((size_t)N * 9);
  float* tb    = alloc((size_t)N * 3);
  float* loc   = alloc((size_t)N * DCH);
  float* posb  = alloc((size_t)N * 42);
  float* feat  = alloc((size_t)N * 266);
  float* xb    = alloc((size_t)N * DCH);
  float* qb    = alloc((size_t)N * DCH);
  float* kbuf  = alloc((size_t)N * DCH);
  float* vbuf  = alloc((size_t)N * DCH);
  float* qp    = alloc((size_t)N * 96);
  float* kp    = alloc((size_t)N * 96);
  float* vp    = alloc((size_t)N * 192);
  float* qpg   = alloc((size_t)N * 96);
  float* kpg   = alloc((size_t)N * 96);
  float* vpg   = alloc((size_t)N * 192);
  float* attc  = alloc((size_t)N * 1024);
  float* t1    = alloc((size_t)N * 512);
  float* t2    = alloc((size_t)N * 512);
  float* upd   = alloc((size_t)N * 42);
  float* pairb = alloc((size_t)N * KNB * PCH);
  int*   aaty  = (int*)alloc((size_t)N);

  // Final-phase aliases (buffers free after the layer loop):
  float* feat2 = attc;   // 346 <= 1024
  float* llb   = kp;     // 42 <= 96
  float* x2    = kbuf;   // 256
  float* aab   = vbuf;   // 20 <= 256
  float* locF  = qb;     // 256
  float* pb    = upd;    // 42
  float* zb    = attc;   // 298 <= 1024 (feat2 consumed before z is built)
  float* g1    = t1;
  float* g2    = t2;

  auto gemm = [&](int M, int Nn, int Kk, const float* A, int lda,
                  const float* B, int ldb, float* C, int ldc,
                  const float* bias, const float* mul, int beta, int act) {
    dim3 g((Nn + TN - 1) / TN, (M + TM - 1) / TM), blk(128);
    gemm_bf16_wmma<<<g, blk, 0, stream>>>(M, Nn, Kk, A, lda, B, ldb, C, ldc,
                                          bias, mul, beta, act);
  };

  hipMemcpyAsync(loc,  in_local, (size_t)N * DCH * sizeof(float),
                 hipMemcpyDeviceToDevice, stream);
  hipMemcpyAsync(posb, in_pos,   (size_t)N * 42 * sizeof(float),
                 hipMemcpyDeviceToDevice, stream);

  // ================= layer loop =================
  for (int p = 0; p < NLAY; ++p) {
    frames_kernel<<<N, 64, 0, stream>>>(posb, Rb, tb, feat, 0);
    gemm(N, DCH, 266, feat, 266, w_feat + (size_t)p * 266 * DCH, DCH,
         loc, DCH, nullptr, nullptr, 1, 0);
    pair_kernel<<<N, 256, 0, stream>>>(Rb, tb, nbr, resi, chain,
                                       w_pair + (size_t)p * 85 * PCH, pairb);
    ln_kernel<<<N, 256, 0, stream>>>(loc, xb);
    gemm(N, DCH, DCH, xb, DCH, w_q + (size_t)p * DCH * DCH, DCH, qb, DCH,
         nullptr, nullptr, 0, 0);
    gemm(N, DCH, DCH, xb, DCH, w_k + (size_t)p * DCH * DCH, DCH, kbuf, DCH,
         nullptr, nullptr, 0, 0);
    gemm(N, DCH, DCH, xb, DCH, w_v + (size_t)p * DCH * DCH, DCH, vbuf, DCH,
         nullptr, nullptr, 0, 0);
    gemm(N, 96, DCH, xb, DCH, w_qp + (size_t)p * DCH * 96, 96, qp, 96,
         nullptr, nullptr, 0, 0);
    gemm(N, 96, DCH, xb, DCH, w_kp + (size_t)p * DCH * 96, 96, kp, 96,
         nullptr, nullptr, 0, 0);
    gemm(N, 192, DCH, xb, DCH, w_vp + (size_t)p * DCH * 192, 192, vp, 192,
         nullptr, nullptr, 0, 0);
    ptrans_kernel<<<N, 192, 0, stream>>>(Rb, tb, qp, kp, vp, qpg, kpg, vpg);
    attn_kernel<<<N, 256, 0, stream>>>(nbr, qb, kbuf, vbuf, qpg, kpg, vpg,
                                       pairb, w_bias + (size_t)p * PCH * NH,
                                       gamma + (size_t)p * NH, Rb, tb, attc);
    gemm(N, DCH, 1024, attc, 1024, w_ao + (size_t)p * 1024 * DCH, DCH,
         loc, DCH, nullptr, nullptr, 1, 0);
    ln_kernel<<<N, 256, 0, stream>>>(loc, xb);
    gemm(N, 512, DCH, xb, DCH, w_g  + (size_t)p * DCH * 512, 512, t1, 512,
         nullptr, nullptr, 0, 1);                      // gelu(y@wg)
    gemm(N, 512, DCH, xb, DCH, w_vm + (size_t)p * DCH * 512, 512, t2, 512,
         nullptr, t1, 0, 0);                           // (y@wvm)*gelu(..)
    gemm(N, DCH, 512, t2, 512, w_mo + (size_t)p * 512 * DCH, DCH, loc, DCH,
         nullptr, nullptr, 1, 0);
    ln_kernel<<<N, 256, 0, stream>>>(loc, xb);
    gemm(N, 42, DCH, xb, DCH, w_pu + (size_t)p * DCH * 42, 42, upd, 42,
         nullptr, nullptr, 0, 0);
    posupd_kernel<<<N, 64, 0, stream>>>(Rb, tb, upd, posb);
  }

  // ================= final head =================
  ln_kernel<<<N, 256, 0, stream>>>(loc, xb);               // local = ln(local)
  gemm(N, 20, DCH, xb, DCH, w_aa, 20, aab, 20, nullptr, nullptr, 0, 0);
  ln_kernel<<<N, 256, 0, stream>>>(xb, x2);                // ln(local) for feat
  frames_kernel<<<N, 64, 0, stream>>>(posb, Rb, tb, llb, 1);
  aa_kernel<<<N, 32, 0, stream>>>(aab, x2, llb, feat2, aaty, out_aa, out_mask);
  gemm(N, 512, 346, feat2, 346, w_m1, 512, t1, 512, b_m1, nullptr, 0, 1);
  gemm(N, DCH, 512, t1, 512, w_m2, DCH, t2, DCH, b_m2, nullptr, 0, 0);
  ln_kernel<<<N, 256, 0, stream>>>(t2, locF);
  gemm(N, 42, DCH, locF, DCH, w_init, 42, pb, 42, b_init, nullptr, 0, 0);
  p4_kernel<<<N, 32, 0, stream>>>(llb, pb);
  for (int it = 0; it < 3; ++it) {
    zcat_kernel<<<N, 128, 0, stream>>>(locF, pb, zb);
    gemm(N, 512, 298, zb, 298, w_ug, 512, g1, 512, nullptr, nullptr, 0, 1);
    gemm(N, 512, 298, zb, 298, w_uv, 512, g2, 512, nullptr, g1, 0, 0);
    gemm(N, 42, 512, g2, 512, w_uo, 42, pb, 42, nullptr, nullptr, 1, 0);
  }
  posout_kernel<<<N, 64, 0, stream>>>(Rb, tb, pb, out_pos);
}
